// PPOLSTMCommAgent_15582141350574
// MI455X (gfx1250) — matrix operand order repaired and checked by reading, not compiled, not run on hardware
//
#include <hip/hip_runtime.h>
#include <hip/hip_bf16.h>

// ---------------------------------------------------------------------------
// PPO LSTM Comm Agent — CDNA5 (gfx1250) implementation.
//   T=1024, B=256, N=262144, D=128, IN=36 (padded to 64), gates=512.
//   GEMMs via v_wmma_f32_16x16x32_f16 (wave32 WMMA).
//   LSTM feat stream double-buffered in LDS via global_load_async_to_lds_b128
//   (ASYNCcnt) so the serial recurrence never waits on HBM.
// ---------------------------------------------------------------------------

#define T_STEPS 1024
#define BATCH   256
#define NROWS   (T_STEPS * BATCH)
#define DMODEL  128
#define KFEAT   64          // 36 padded to 64
#define KGATE   192         // 64 (x) + 128 (h)
#define NGATE   512

typedef __attribute__((ext_vector_type(16))) _Float16 v16h;
typedef __attribute__((ext_vector_type(8)))  _Float16 v8h;
typedef __attribute__((ext_vector_type(8)))  float    v8f;

// ---------------- WMMA fragment helpers (16x16x32 f16, wave32) -------------
// A (16x32): lanes 0-15 -> M=0..15, K {0..7, 16..23}; lanes 16-31 same M,
//            K {8..15, 24..31}. Two contiguous 16B loads per lane.
__device__ __forceinline__ v16h frag_a(const _Float16* p /*[m0][k0]*/, int lda, int lane) {
  int m  = lane & 15;
  int hi = lane >> 4;
  const _Float16* q = p + m * lda + hi * 8;
  v8h lo = *(const v8h*)(q);
  v8h hh = *(const v8h*)(q + 16);
  return __builtin_shufflevector(lo, hh, 0,1,2,3,4,5,6,7,8,9,10,11,12,13,14,15);
}

// B (32x16) from row-major W[n][k] (== B^T): lane n=lane&15, same K order as
// frag_a: lanes 0-15 K {0..7, 16..23}, lanes 16-31 K {8..15, 24..31}.
__device__ __forceinline__ v16h frag_b(const _Float16* p /*[n0][k0]*/, int ldb, int lane) {
  int n  = lane & 15;
  int hi = lane >> 4;
  const _Float16* q = p + n * ldb + hi * 8;
  v8h lo = *(const v8h*)(q);
  v8h hh = *(const v8h*)(q + 16);
  return __builtin_shufflevector(lo, hh, 0,1,2,3,4,5,6,7,8,9,10,11,12,13,14,15);
}

__device__ __forceinline__ v8f wmma16(v16h a, v16h b, v8f c) {
  return __builtin_amdgcn_wmma_f32_16x16x32_f16(false, a, false, b, (short)0, c,
                                                false, false);
}

__device__ __forceinline__ float fsig(float x) {
  return 1.0f / (1.0f + __expf(-x));
}
__device__ __forceinline__ float ftanh(float x) {
  return 2.0f * fsig(2.0f * x) - 1.0f;
}

// ---------------- weight prep kernels --------------------------------------
__global__ void k_pad_convert(const float* __restrict__ src, _Float16* __restrict__ dst,
                              int Nr, int K, int Kp) {
  int i = blockIdx.x * 256 + threadIdx.x;
  if (i >= Nr * Kp) return;
  int n = i / Kp, k = i - n * Kp;
  dst[i] = (_Float16)((k < K) ? src[n * K + k] : 0.0f);
}

// Wg[512][192]: cols 0..35 = W_ih, 36..63 = 0, 64..191 = W_hh
__global__ void k_build_wg(const float* __restrict__ wih, const float* __restrict__ whh,
                           _Float16* __restrict__ dst) {
  int i = blockIdx.x * 256 + threadIdx.x;
  if (i >= NGATE * KGATE) return;
  int n = i / KGATE, k = i - n * KGATE;
  float v = 0.0f;
  if (k < 36)       v = wih[n * 36 + k];
  else if (k >= 64) v = whh[n * 128 + (k - 64)];
  dst[i] = (_Float16)v;
}

// Whead[32][128]: rows 0..4 actor, 5..20 head, 21 critic, 22..31 zero.
// bg = b_ih + b_hh ; bh = concat biases.
__global__ void k_build_heads(const float* __restrict__ aw, const float* __restrict__ hw,
                              const float* __restrict__ cw, const float* __restrict__ ab,
                              const float* __restrict__ hb, const float* __restrict__ cb,
                              const float* __restrict__ bih, const float* __restrict__ bhh,
                              _Float16* __restrict__ wdst, float* __restrict__ bhdst,
                              float* __restrict__ bgdst) {
  int i = blockIdx.x * 256 + threadIdx.x;
  if (i < 32 * 128) {
    int n = i >> 7, k = i & 127;
    float v = 0.0f;
    if (n < 5)       v = aw[n * 128 + k];
    else if (n < 21) v = hw[(n - 5) * 128 + k];
    else if (n < 22) v = cw[k];
    wdst[i] = (_Float16)v;
  }
  if (i < NGATE) bgdst[i] = bih[i] + bhh[i];
  if (i < 32) {
    float v = 0.0f;
    if (i < 5)       v = ab[i];
    else if (i < 21) v = hb[i - 5];
    else if (i < 22) v = cb[0];
    bhdst[i] = v;
  }
}

// ---------------- encoder: fused 4-layer MLP + feat assembly ---------------
// One block = 64 rows (4 M-tiles), 256 threads = 8 waves. Activations ping-
// pong in LDS with fixed row stride 256 halves; weights (f16, row-major
// [N][Kp]) are read straight from L2 via fragment loads.
__device__ __forceinline__ void mlp_layer(const _Float16* sIn, _Float16* sOut,
                                          const _Float16* __restrict__ W,
                                          const float* __restrict__ bias,
                                          int Kp, int Nout, int lane, int wave) {
  int ktiles = Kp >> 5;
  int ntot   = (Nout >> 4) * 4;         // 4 M-tiles x N-tiles
  for (int ct = wave; ct < ntot; ct += 8) {
    int mt = ct & 3, nt = ct >> 2;
    v8f acc = {};
    for (int kt = 0; kt < ktiles; ++kt) {
      v16h a = frag_a(sIn + (mt * 16) * 256 + kt * 32, 256, lane);
      v16h b = frag_b(W + (nt * 16) * Kp + kt * 32, Kp, lane);
      acc = wmma16(a, b, acc);
    }
    int   n  = nt * 16 + (lane & 15);
    float bv = bias[n];
    int   mb = mt * 16 + ((lane >> 4) << 3);
#pragma unroll
    for (int r = 0; r < 8; ++r) {
      float v = fmaxf(acc[r] + bv, 0.0f);
      sOut[(mb + r) * 256 + n] = (_Float16)v;
    }
  }
}

__global__ void k_encoder(const float* __restrict__ image,
                          const float* __restrict__ location,
                          const int*   __restrict__ message,
                          const float* __restrict__ emb,
                          const float* __restrict__ msg_W,
                          const float* __restrict__ msg_b,
                          const float* __restrict__ loc_W,
                          const float* __restrict__ loc_b,
                          const _Float16* __restrict__ W1p, const float* __restrict__ b1,
                          const _Float16* __restrict__ W2p, const float* __restrict__ b2,
                          const _Float16* __restrict__ W3p, const float* __restrict__ b3,
                          const _Float16* __restrict__ W4p, const float* __restrict__ b4,
                          _Float16* __restrict__ featg) {
  extern __shared__ _Float16 smem[];
  _Float16* sA = smem;               // 64 x 256 halves
  _Float16* sB = smem + 64 * 256;    // 64 x 256 halves
  int tid = threadIdx.x, lane = tid & 31, wave = tid >> 5;
  long row0 = (long)blockIdx.x * 64;

  // stage image/255 into padded [64][32] f16
  for (int idx = tid; idx < 64 * 32; idx += 256) {
    int m = idx >> 5, k = idx & 31;
    float v = (k < 9) ? image[(row0 + m) * 9 + k] * (1.0f / 255.0f) : 0.0f;
    sA[m * 256 + k] = (_Float16)v;
  }
  __syncthreads();
  mlp_layer(sA, sB, W1p, b1,  32, 256, lane, wave); __syncthreads();
  mlp_layer(sB, sA, W2p, b2, 256, 256, lane, wave); __syncthreads();
  mlp_layer(sA, sB, W3p, b3, 256, 128, lane, wave); __syncthreads();
  mlp_layer(sB, sA, W4p, b4, 128,  16, lane, wave); __syncthreads();

  // feat = [img16 | loc4 | msg16 | pad] -> global f16 [N][64]
  for (int idx = tid; idx < 64 * 64; idx += 256) {
    int m = idx >> 6, j = idx & 63;
    long row = row0 + m;
    float v;
    if (j < 16) {
      v = (float)sA[m * 256 + j];
    } else if (j < 20) {
      int jj = j - 16;
      v = (location[row * 2 + 0] * 0.2f) * loc_W[jj * 2 + 0]
        + (location[row * 2 + 1] * 0.2f) * loc_W[jj * 2 + 1] + loc_b[jj];
    } else if (j < 36) {
      int jj  = j - 20;
      int msg = message[row];
      float acc = msg_b[jj];
#pragma unroll
      for (int k = 0; k < 16; ++k) acc += emb[msg * 16 + k] * msg_W[jj * 16 + k];
      v = fmaxf(acc, 0.0f);
    } else {
      v = 0.0f;
    }
    featg[row * 64 + j] = (_Float16)v;
  }
}

// ---------------- persistent LSTM + heads ----------------------------------
// 16 workgroups, each owns 16 batch rows for all 1024 steps. Wg[512][192]
// (192KB f16) + head weights + state live in the 320KB WGP LDS. The per-step
// feat slice (2KB) is double-buffered in LDS via async global->LDS DMA
// (ASYNCcnt), issued one step ahead by wave 0 and drained with
// s_wait_asynccnt before the end-of-step barrier.

// Issue 2KB (16 rows x 128B) of feat for `baseNext` into sFeatBuf.
// One wave: 4 x b128 per lane. LDS operand = low 32 bits of generic pointer
__device__ __forceinline__ void async_feat_issue(const _Float16* __restrict__ featg,
                                                 long baseNext, _Float16* sFeatBuf,
                                                 int lane) {
  unsigned ldsBase = (unsigned)(unsigned long long)sFeatBuf;
#pragma unroll
  for (int j = 0; j < 4; ++j) {
    int c = lane * 4 + j;              // 128 chunks of 16B
    int r = c >> 3;
    int o = (c & 7) * 16;
    unsigned gOff = (unsigned)((baseNext + r) * (KFEAT * 2)) + (unsigned)o;
    unsigned lOff = ldsBase + (unsigned)(r * (KFEAT * 2) + o);
    asm volatile("global_load_async_to_lds_b128 %0, %1, %2"
                 :: "v"(lOff), "v"(gOff), "s"(featg)
                 : "memory");
  }
}

__device__ __forceinline__ void async_drain() {
  asm volatile("s_wait_asynccnt 0x0" ::: "memory");
}

__global__ void k_lstm(const _Float16* __restrict__ featg,
                       const float* __restrict__ done,
                       const float* __restrict__ h0,
                       const float* __restrict__ c0,
                       const _Float16* __restrict__ Wg,  const float* __restrict__ bg,
                       const _Float16* __restrict__ Whd, const float* __restrict__ bhd,
                       float* __restrict__ out) {
  extern __shared__ char smemraw[];
  _Float16* sWg = (_Float16*)smemraw;            // 512*192
  _Float16* sWh = sWg + NGATE * KGATE;           // 32*128
  _Float16* sH  = sWh + 32 * 128;                // 16*128 f16 state
  float*    sC  = (float*)(sH + 16 * 128);       // 16*128 f32 state
  float*    sG  = sC + 16 * 128;                 // 16*512 gates
  float*    sBg = sG + 16 * NGATE;               // 512
  float*    sBh = sBg + NGATE;                   // 32
  __shared__ _Float16 sFeat[2][16 * KFEAT];      // feat ping-pong (2 x 2KB)

  int tid = threadIdx.x, lane = tid & 31, wave = tid >> 5;
  int b0  = blockIdx.x * 16;

  { // resident weights -> LDS (16B moves)
    const uint4* src = (const uint4*)Wg; uint4* dst = (uint4*)sWg;
    for (int i = tid; i < NGATE * KGATE * 2 / 16; i += 256) dst[i] = src[i];
  }
  { const uint4* src = (const uint4*)Whd; uint4* dst = (uint4*)sWh;
    for (int i = tid; i < 32 * 128 * 2 / 16; i += 256) dst[i] = src[i];
  }
  for (int i = tid; i < NGATE; i += 256) sBg[i] = bg[i];
  if (tid < 32) sBh[tid] = bhd[tid];
  for (int i = tid; i < 2048; i += 256) {
    int r = i >> 7, d = i & 127;
    sH[i] = (_Float16)h0[(b0 + r) * 128 + d];
    sC[i] = c0[(b0 + r) * 128 + d];
  }
  if (wave == 0) {                     // prime feat buffer 0 for t=0
    async_feat_issue(featg, (long)b0, &sFeat[0][0], lane);
    async_drain();
  }
  __syncthreads();

  for (int t = 0; t < T_STEPS; ++t) {
    long base = (long)t * BATCH + b0;

    // kick off async DMA of next step's feat slice (overlaps whole step)
    if (wave == 0 && t + 1 < T_STEPS)
      async_feat_issue(featg, base + BATCH, &sFeat[(t + 1) & 1][0], lane);
    if (t + 2 < T_STEPS)               // warm L2 two steps ahead
      __builtin_prefetch((const void*)(featg + (base + 2 * BATCH) * KFEAT + (long)tid * 8), 0, 0);

    // done-mask state
    for (int i = tid; i < 2048; i += 256) {
      int r = i >> 7;
      float m = 1.0f - done[base + r];
      sH[i] = (_Float16)((float)sH[i] * m);
      sC[i] *= m;
    }
    __syncthreads();

    const _Float16* featT = &sFeat[t & 1][0];

    // gates[16][512] = [feat|h][16][192] x Wg^T, bias fused
#pragma unroll
    for (int ni = 0; ni < 4; ++ni) {
      int nt = wave * 4 + ni;
      v8f acc = {};
#pragma unroll
      for (int kt = 0; kt < 2; ++kt) {          // x part (LDS feat buffer)
        v16h a = frag_a(featT + kt * 32, KFEAT, lane);
        v16h b = frag_b(sWg + nt * 16 * KGATE + kt * 32, KGATE, lane);
        acc = wmma16(a, b, acc);
      }
#pragma unroll
      for (int kt = 0; kt < 4; ++kt) {          // h part (LDS state)
        v16h a = frag_a(sH + kt * 32, 128, lane);
        v16h b = frag_b(sWg + nt * 16 * KGATE + 64 + kt * 32, KGATE, lane);
        acc = wmma16(a, b, acc);
      }
      int   n  = nt * 16 + (lane & 15);
      float bv = sBg[n];
      int   mb = (lane >> 4) << 3;
#pragma unroll
      for (int r = 0; r < 8; ++r) sG[(mb + r) * NGATE + n] = acc[r] + bv;
    }
    __syncthreads();

    // pointwise cell (gate order i,f,g,o)
    for (int i = tid; i < 2048; i += 256) {
      int r = i >> 7, d = i & 127;
      const float* g = sG + r * NGATE;
      float ig = fsig(g[d]);
      float fg = fsig(g[128 + d]);
      float gg = ftanh(g[256 + d]);
      float og = fsig(g[384 + d]);
      float c2 = fg * sC[i] + ig * gg;
      float h2 = og * ftanh(c2);
      sC[i] = c2;
      sH[i] = (_Float16)h2;
    }
    __syncthreads();

    // heads: [16][128] x Whead[32][128]^T -> 22 logits/row
    if (wave < 2) {
      v8f acc = {};
#pragma unroll
      for (int kt = 0; kt < 4; ++kt) {
        v16h a = frag_a(sH + kt * 32, 128, lane);
        v16h b = frag_b(sWh + wave * 16 * 128 + kt * 32, 128, lane);
        acc = wmma16(a, b, acc);
      }
      int n = wave * 16 + (lane & 15);
      if (n < 22) {
        float bv = sBh[n];
        int   mb = (lane >> 4) << 3;
#pragma unroll
        for (int r = 0; r < 8; ++r)
          out[(base + mb + r) * 22 + n] = acc[r] + bv;
      }
    }
    if (wave == 0) async_drain();      // next-step feat resident before barrier
    __syncthreads();                   // also protects sH before next masking
  }
}

// ---------------- launch ----------------------------------------------------
extern "C" void kernel_launch(void* const* d_in, const int* in_sizes, int n_in,
                              void* d_out, int out_size, void* d_ws, size_t ws_size,
                              hipStream_t stream) {
  const float* image    = (const float*)d_in[0];
  const float* location = (const float*)d_in[1];
  const int*   message  = (const int*)  d_in[2];
  const float* done     = (const float*)d_in[3];
  const float* h0       = (const float*)d_in[4];
  const float* c0       = (const float*)d_in[5];
  const float* vis_W1   = (const float*)d_in[6];
  const float* vis_b1   = (const float*)d_in[7];
  const float* vis_W2   = (const float*)d_in[8];
  const float* vis_b2   = (const float*)d_in[9];
  const float* vis_W3   = (const float*)d_in[10];
  const float* vis_b3   = (const float*)d_in[11];
  const float* vis_W4   = (const float*)d_in[12];
  const float* vis_b4   = (const float*)d_in[13];
  const float* loc_W    = (const float*)d_in[14];
  const float* loc_b    = (const float*)d_in[15];
  const float* emb      = (const float*)d_in[16];
  const float* msg_W    = (const float*)d_in[17];
  const float* msg_b    = (const float*)d_in[18];
  const float* W_ih     = (const float*)d_in[19];
  const float* W_hh     = (const float*)d_in[20];
  const float* b_ih     = (const float*)d_in[21];
  const float* b_hh     = (const float*)d_in[22];
  const float* actor_W  = (const float*)d_in[23];
  const float* actor_b  = (const float*)d_in[24];
  const float* head_W   = (const float*)d_in[25];
  const float* head_b   = (const float*)d_in[26];
  const float* critic_W = (const float*)d_in[27];
  const float* critic_b = (const float*)d_in[28];
  float* out = (float*)d_out;

  // workspace layout (256B aligned)
  char* ws = (char*)d_ws;
  _Float16* W1p   = (_Float16*)(ws + 0);           // 256x32
  _Float16* W2p   = (_Float16*)(ws + 16384);       // 256x256
  _Float16* W3p   = (_Float16*)(ws + 147456);      // 128x256
  _Float16* W4p   = (_Float16*)(ws + 212992);      // 16x128
  _Float16* Wg    = (_Float16*)(ws + 217088);      // 512x192
  _Float16* Whd   = (_Float16*)(ws + 413696);      // 32x128
  float*    bg    = (float*)   (ws + 421888);      // 512
  float*    bhd   = (float*)   (ws + 423936);      // 32
  _Float16* featg = (_Float16*)(ws + 424192);      // N x 64

  // ---- weight prep ----
  k_pad_convert<<<(256 * 32  + 255) / 256, 256, 0, stream>>>(vis_W1, W1p, 256,   9,  32);
  k_pad_convert<<<(256 * 256 + 255) / 256, 256, 0, stream>>>(vis_W2, W2p, 256, 256, 256);
  k_pad_convert<<<(128 * 256 + 255) / 256, 256, 0, stream>>>(vis_W3, W3p, 128, 256, 256);
  k_pad_convert<<<(16  * 128 + 255) / 256, 256, 0, stream>>>(vis_W4, W4p,  16, 128, 128);
  k_build_wg<<<(NGATE * KGATE + 255) / 256, 256, 0, stream>>>(W_ih, W_hh, Wg);
  k_build_heads<<<(32 * 128 + 255) / 256, 256, 0, stream>>>(
      actor_W, head_W, critic_W, actor_b, head_b, critic_b, b_ih, b_hh, Whd, bhd, bg);

  // ---- fused encoder (4096 blocks x 64 rows), 64KB dynamic LDS ----
  size_t enc_lds = 2u * 64 * 256 * sizeof(_Float16);
  k_encoder<<<NROWS / 64, 256, enc_lds, stream>>>(
      image, location, message, emb, msg_W, msg_b, loc_W, loc_b,
      W1p, vis_b1, W2p, vis_b2, W3p, vis_b3, W4p, vis_b4, featg);

  // ---- persistent LSTM + heads (16 blocks, ~246KB dyn LDS + 4KB static) ----
  size_t lstm_lds = (size_t)NGATE * KGATE * 2    // sWg
                  + 32 * 128 * 2                 // sWh
                  + 16 * 128 * 2                 // sH
                  + 16 * 128 * 4                 // sC
                  + 16 * NGATE * 4               // sG
                  + NGATE * 4 + 32 * 4;          // biases
  k_lstm<<<BATCH / 16, 256, lstm_lds, stream>>>(
      featg, done, h0, c0, Wg, bg, Whd, bhd, out);

  (void)in_sizes; (void)n_in; (void)out_size; (void)ws_size;
}
